// GCN_3607772529053
// MI455X (gfx1250) — matrix operand order repaired and checked by reading, not compiled, not run on hardware
//
#include <hip/hip_runtime.h>
#include <math.h>

// ---------------------------------------------------------------------------
// GCN on MI455X (gfx1250, wave32). Dense projections use
// v_wmma_f32_16x16x32_f16 with a pre-packed transposed-f16 weight buffer so
// the B fragment is one aligned 32B load. Edge aggregation uses float4
// gathers + global_atomic_add_f32 (feature matrix is 25MB << 192MB L2).
// ---------------------------------------------------------------------------

typedef __attribute__((ext_vector_type(16))) _Float16 v16h;
typedef __attribute__((ext_vector_type(8)))  float    v8f;

#define DKH 64        // hidden dim
#define GG  256       // graphs
#define SLOPE 0.01f
#define BN_EPS 1e-5f

// ---------------------------------------------------------------- utilities
__global__ void fill_kernel(float* __restrict__ p, float v, int n) {
  int i = blockIdx.x * blockDim.x + threadIdx.x;
  if (i < n) p[i] = v;
}

__device__ __forceinline__ void atomicMaxFloat(float* addr, float v) {
  // int ordering matches float ordering for non-negatives; reversed (as uint)
  // for negatives. Init value must be -inf (0xFF800000).
  if (v >= 0.0f) atomicMax((int*)addr, __float_as_int(v));
  else           atomicMin((unsigned int*)addr, __float_as_uint(v));
}

// ------------------------------------------------------------- degree/dinv
__global__ void deg_init_kernel(float* __restrict__ deg, int n) {
  int i = blockIdx.x * blockDim.x + threadIdx.x;
  if (i < n) deg[i] = 1.0f;                    // self-loop weight
}

__global__ void deg_acc_kernel(const int* __restrict__ dst,
                               const float* __restrict__ w,
                               float* __restrict__ deg, int e) {
  int i = blockIdx.x * blockDim.x + threadIdx.x;
  if (i < e) atomicAdd(&deg[dst[i]], w[i]);
}

__global__ void deg_finish_kernel(float* __restrict__ deg, int n) {
  int i = blockIdx.x * blockDim.x + threadIdx.x;
  if (i < n) {
    float d = deg[i];
    deg[i] = (d > 0.0f) ? rsqrtf(fmaxf(d, 1e-12f)) : 0.0f;   // becomes dinv
  }
}

// -------------------------------------------------- weight pack (f32->f16^T)
// W[K,64] row-major f32  ->  Wt[64,K] f16 (column of W contiguous).
__global__ void pack_w_kernel(const float* __restrict__ W,
                              _Float16* __restrict__ Wt, int K) {
  int idx = blockIdx.x * blockDim.x + threadIdx.x;
  if (idx >= K * DKH) return;
  const int k = idx / DKH, c = idx % DKH;
  Wt[c * K + k] = (_Float16)W[idx];
}

// ------------------------------------------------------------ WMMA GEMM
// Out[N,64] = X[N,K] @ W[K,64].  One block = 16 rows; 4 waves cover the 64
// output columns (16 each).  f16 inputs, f32 accumulate.
//
// A-frag (16x32 f16, contiguous): lanes 0-15 row=lane hold
//   K 0..15; lanes 16-31 same rows hold K 16..31.
// B-frag (32x16 f16, contiguous, same k order as A): lane = col,
//   lanes 0-15 hold K 0..15 in elems 0..15; lanes 16-31 hold K 16..31.
template <int K>
__global__ __launch_bounds__(128)
void gemm_wmma_kernel(const float* __restrict__ X,
                      const _Float16* __restrict__ Wt,
                      float* __restrict__ Out, int nrows) {
  const int m0   = blockIdx.x * 16;
  const int wave = threadIdx.x >> 5;
  const int lane = threadIdx.x & 31;
  const int hh   = lane >> 4;        // half-wave: 0 or 1
  const int r    = lane & 15;        // row (A) / col (B) within tile
  const int n0   = wave * 16;

  int arowIdx = m0 + r;
  if (arowIdx >= nrows) arowIdx = nrows - 1;   // clamp (guarded on store)
  const float*    arow = X + (size_t)arowIdx * K;
  const _Float16* bcol = Wt + (size_t)(n0 + r) * K;

  v8f c = {};
#pragma unroll
  for (int kk = 0; kk < K; kk += 32) {
    const int kb = kk + hh * 16;
    v16h a;
#pragma unroll
    for (int e = 0; e < 8; ++e) {
      a[e]     = (_Float16)arow[kb + e];
      a[e + 8] = (_Float16)arow[kb + 8 + e];
    }
    // contiguous 16 halves = one aligned 32B load
    const v16h b = *reinterpret_cast<const v16h*>(bcol + kk + hh * 16);
    // D = A*B + C   (f32 accumulator chain)
    c = __builtin_amdgcn_wmma_f32_16x16x32_f16(false, a, false, b,
                                               (short)0, c, false, false);
  }

  // C/D layout: lane half 0 -> rows 0..7, half 1 -> rows 8..15; col = n0+r.
#pragma unroll
  for (int rr = 0; rr < 8; ++rr) {
    const int row = m0 + hh * 8 + rr;
    if (row < nrows) Out[(size_t)row * DKH + n0 + r] = c[rr];
  }
}

// -------------------------------------------------------- aggregation
// Out[n] = T[n] * dinv[n]^2  (self loop, weight 1)
__global__ void agg_init_kernel(const float* __restrict__ T,
                                const float* __restrict__ dinv,
                                float* __restrict__ Out, int n) {
  int idx = blockIdx.x * blockDim.x + threadIdx.x;
  if (idx >= n * DKH) return;
  const int node = idx >> 6;
  const float di = dinv[node];
  Out[idx] = T[idx] * di * di;
}

// One thread per (edge, 4-channel chunk): float4 gather + 4 f32 atomics.
__global__ void agg_edge_kernel(const int* __restrict__ src,
                                const int* __restrict__ dst,
                                const float* __restrict__ w,
                                const float* __restrict__ dinv,
                                const float* __restrict__ T,
                                float* __restrict__ Out, int e) {
  const int idx = blockIdx.x * blockDim.x + threadIdx.x;
  const int ei  = idx >> 4;
  if (ei >= e) return;
  const int c4 = (idx & 15) << 2;
  const int s = src[ei], d = dst[ei];
  const float coef = dinv[s] * w[ei] * dinv[d];
  const float4 v = *reinterpret_cast<const float4*>(T + (size_t)s * DKH + c4);
  float* o = Out + (size_t)d * DKH + c4;
  atomicAdd(o + 0, v.x * coef);
  atomicAdd(o + 1, v.y * coef);
  atomicAdd(o + 2, v.z * coef);
  atomicAdd(o + 3, v.w * coef);
}

// ------------------------------------------------------------- batch norm
__global__ __launch_bounds__(256)
void bn_stats_kernel(const float* __restrict__ H, float* __restrict__ stats,
                     int n) {
  __shared__ float ls[2 * DKH];
  if (threadIdx.x < 2 * DKH) ls[threadIdx.x] = 0.0f;
  __syncthreads();
  const int c = threadIdx.x & 63;
  float s = 0.0f, sq = 0.0f;
  for (int row = blockIdx.x * 4 + (threadIdx.x >> 6); row < n;
       row += gridDim.x * 4) {
    const float v = H[(size_t)row * DKH + c];
    s += v; sq += v * v;
  }
  atomicAdd(&ls[c], s);
  atomicAdd(&ls[DKH + c], sq);
  __syncthreads();
  if (threadIdx.x < 2 * DKH) atomicAdd(&stats[threadIdx.x], ls[threadIdx.x]);
}

// BN + LeakyReLU in place. The conv bias shifts mean identically, so it
// cancels inside (h - mean) and is omitted exactly.
__global__ void bn_apply_lrelu_kernel(float* __restrict__ H,
                                      const float* __restrict__ stats,
                                      const float* __restrict__ gamma,
                                      const float* __restrict__ beta, int n) {
  const int idx = blockIdx.x * blockDim.x + threadIdx.x;
  if (idx >= n * DKH) return;
  const int c = idx & 63;
  const float invn = 1.0f / (float)n;
  const float mean = stats[c] * invn;
  const float var  = stats[DKH + c] * invn - mean * mean;
  const float o = (H[idx] - mean) * rsqrtf(var + BN_EPS) * gamma[c] + beta[c];
  H[idx] = (o > 0.0f) ? o : SLOPE * o;
}

__global__ void bias_lrelu_kernel(float* __restrict__ H,
                                  const float* __restrict__ b, int n) {
  const int idx = blockIdx.x * blockDim.x + threadIdx.x;
  if (idx >= n * DKH) return;
  const float o = H[idx] + b[idx & 63];
  H[idx] = (o > 0.0f) ? o : SLOPE * o;
}

// --------------------------------------------------------------- pooling
__global__ void pool_acc_kernel(const float* __restrict__ H,
                                const int* __restrict__ batch,
                                float* __restrict__ mx, float* __restrict__ sm,
                                float* __restrict__ cnt, int n) {
  const int idx = blockIdx.x * blockDim.x + threadIdx.x;
  if (idx >= n * DKH) return;
  const int node = idx >> 6, c = idx & 63;
  const int g = batch[node];
  const float v = H[idx];
  atomicMaxFloat(&mx[g * DKH + c], v);
  atomicAdd(&sm[g * DKH + c], v);
  if (c == 0) atomicAdd(&cnt[g], 1.0f);
}

__global__ void fc_kernel(const float* __restrict__ mx,
                          const float* __restrict__ sm,
                          const float* __restrict__ cnt,
                          const float* __restrict__ Wfc,
                          const float* __restrict__ bfc,
                          float* __restrict__ out) {
  const int g = blockIdx.x * blockDim.x + threadIdx.x;
  if (g >= GG) return;
  const float inv = 1.0f / fmaxf(cnt[g], 1.0f);
  float acc = bfc[0];
#pragma unroll 8
  for (int d = 0; d < DKH; ++d)
    acc += mx[g * DKH + d] * Wfc[d] + sm[g * DKH + d] * inv * Wfc[DKH + d];
  out[g] = acc;
}

// ---------------------------------------------------------------- driver
extern "C" void kernel_launch(void* const* d_in, const int* in_sizes, int n_in,
                              void* d_out, int out_size, void* d_ws,
                              size_t ws_size, hipStream_t stream) {
  (void)n_in; (void)out_size; (void)ws_size;
  const float* x    = (const float*)d_in[0];   // [N,128]
  const int*   ei   = (const int*)  d_in[1];   // [2,E]
  const int*   bat  = (const int*)  d_in[2];   // [N]
  const float* ea   = (const float*)d_in[3];   // [E]
  const float* W0   = (const float*)d_in[4];
  const float* g0   = (const float*)d_in[6];
  const float* be0  = (const float*)d_in[7];
  const float* W1   = (const float*)d_in[8];
  const float* g1   = (const float*)d_in[10];
  const float* be1  = (const float*)d_in[11];
  const float* W2   = (const float*)d_in[12];
  const float* b2   = (const float*)d_in[13];
  const float* W3   = (const float*)d_in[14];
  const float* b3   = (const float*)d_in[15];
  const float* Wfc  = (const float*)d_in[16];
  const float* bfc  = (const float*)d_in[17];
  float* out = (float*)d_out;

  const int N = in_sizes[2];
  const int E = in_sizes[3];
  const int* src = ei;
  const int* dst = ei + E;

  // ---- workspace carve-up (floats, 32B-aligned regions) ----
  float* ws   = (float*)d_ws;
  float* dinv = ws;                   size_t o = (size_t)N;      o = (o + 63) & ~63ull;
  float* T    = ws + o;               o += (size_t)N * DKH;
  float* H    = ws + o;               o += (size_t)N * DKH;
  float* st0  = ws + o;               o += 2 * DKH;   // BN0 stats
  float* st1  = ws + o;               o += 2 * DKH;   // BN1 stats
  float* pmx  = ws + o;               o += (size_t)GG * DKH;
  float* psm  = ws + o;               o += (size_t)GG * DKH;
  float* pcnt = ws + o;               o += GG;        o = (o + 63) & ~63ull;
  _Float16* Wt = (_Float16*)(ws + o); o += (128 * DKH) / 2;  // packed weights

  const int TPB = 256;
  const int gN   = (N + TPB - 1) / TPB;
  const int gE   = (E + TPB - 1) / TPB;
  const int gNd  = (N * DKH + TPB - 1) / TPB;
  const int gE16 = (E * 16 + TPB - 1) / TPB;
  const int gM   = (N + 15) / 16;
  const int gW128 = (128 * DKH + TPB - 1) / TPB;
  const int gW64  = (64 * DKH + TPB - 1) / TPB;

  // ---- per-call scratch init (deterministic, graph-capture safe) ----
  fill_kernel<<<(4 * DKH + TPB - 1) / TPB, TPB, 0, stream>>>(st0, 0.0f, 4 * DKH);
  fill_kernel<<<(GG * DKH + TPB - 1) / TPB, TPB, 0, stream>>>(pmx, -INFINITY, GG * DKH);
  fill_kernel<<<(GG * DKH + TPB - 1) / TPB, TPB, 0, stream>>>(psm, 0.0f, GG * DKH);
  fill_kernel<<<1, TPB, 0, stream>>>(pcnt, 0.0f, GG);

  // ---- symmetric-norm degree -> dinv ----
  deg_init_kernel<<<gN, TPB, 0, stream>>>(dinv, N);
  deg_acc_kernel<<<gE, TPB, 0, stream>>>(dst, ea, dinv, E);
  deg_finish_kernel<<<gN, TPB, 0, stream>>>(dinv, N);

  // ---- layer 0: [N,128]@[128,64] -> agg -> BN -> lrelu ----
  pack_w_kernel<<<gW128, TPB, 0, stream>>>(W0, Wt, 128);
  gemm_wmma_kernel<128><<<gM, 128, 0, stream>>>(x, Wt, T, N);
  agg_init_kernel<<<gNd, TPB, 0, stream>>>(T, dinv, H, N);
  agg_edge_kernel<<<gE16, TPB, 0, stream>>>(src, dst, ea, dinv, T, H, E);
  bn_stats_kernel<<<512, TPB, 0, stream>>>(H, st0, N);
  bn_apply_lrelu_kernel<<<gNd, TPB, 0, stream>>>(H, st0, g0, be0, N);

  // ---- layer 1: [N,64]@[64,64] -> agg -> BN -> lrelu ----
  pack_w_kernel<<<gW64, TPB, 0, stream>>>(W1, Wt, 64);
  gemm_wmma_kernel<64><<<gM, 128, 0, stream>>>(H, Wt, T, N);
  agg_init_kernel<<<gNd, TPB, 0, stream>>>(T, dinv, H, N);
  agg_edge_kernel<<<gE16, TPB, 0, stream>>>(src, dst, ea, dinv, T, H, E);
  bn_stats_kernel<<<512, TPB, 0, stream>>>(H, st1, N);
  bn_apply_lrelu_kernel<<<gNd, TPB, 0, stream>>>(H, st1, g1, be1, N);

  // ---- layer 2: agg -> +bias -> lrelu ----
  pack_w_kernel<<<gW64, TPB, 0, stream>>>(W2, Wt, 64);
  gemm_wmma_kernel<64><<<gM, 128, 0, stream>>>(H, Wt, T, N);
  agg_init_kernel<<<gNd, TPB, 0, stream>>>(T, dinv, H, N);
  agg_edge_kernel<<<gE16, TPB, 0, stream>>>(src, dst, ea, dinv, T, H, E);
  bias_lrelu_kernel<<<gNd, TPB, 0, stream>>>(H, b2, N);

  // ---- layer 3 ----
  pack_w_kernel<<<gW64, TPB, 0, stream>>>(W3, Wt, 64);
  gemm_wmma_kernel<64><<<gM, 128, 0, stream>>>(H, Wt, T, N);
  agg_init_kernel<<<gNd, TPB, 0, stream>>>(T, dinv, H, N);
  agg_edge_kernel<<<gE16, TPB, 0, stream>>>(src, dst, ea, dinv, T, H, E);
  bias_lrelu_kernel<<<gNd, TPB, 0, stream>>>(H, b3, N);

  // ---- pooling (max || mean) + FC ----
  pool_acc_kernel<<<gNd, TPB, 0, stream>>>(H, bat, pmx, psm, pcnt, N);
  fc_kernel<<<1, TPB, 0, stream>>>(pmx, psm, pcnt, Wfc, bfc, out);
}
